// GraphAttentionNetwork_68994354642984
// MI455X (gfx1250) — matrix operand order repaired and checked by reading, not compiled, not run on hardware
//
#include <hip/hip_runtime.h>
#include <math.h>

#define N_NODES 50000
#define N_EDGES 800000
#define IN_F    15
#define HID     64
#define OUT_FC  128
#define HEADS   3
#define NEG_SLOPE 0.15f
#define K1PAD   32
#define K2PAD   224   // 207 padded to 7*32
#define D2IN    207

typedef __attribute__((ext_vector_type(16))) _Float16 v16h;
typedef __attribute__((ext_vector_type(8)))  _Float16 v8h;
typedef __attribute__((ext_vector_type(8)))  float    v8f;

// ---------------- utility ----------------

__global__ void fill_f32(float* p, int n, float v) {
    int i = blockIdx.x * blockDim.x + threadIdx.x;
    if (i < n) p[i] = v;
}

__device__ __forceinline__ unsigned f2o(float f) {
    unsigned b = __float_as_uint(f);
    return (b & 0x80000000u) ? ~b : (b | 0x80000000u);
}
__device__ __forceinline__ float o2f(unsigned u) {
    unsigned b = (u & 0x80000000u) ? (u & 0x7fffffffu) : ~u;
    return __uint_as_float(b);
}

// ---------------- packing to f16 ----------------

// x [N,15] f32 -> A [N,32] f16 zero-padded
__global__ void pack_a1(const float* __restrict__ x, _Float16* __restrict__ A, int n) {
    int i = blockIdx.x * blockDim.x + threadIdx.x;
    if (i >= n) return;
    int node = i >> 5, k = i & 31;
    A[i] = (k < IN_F) ? (_Float16)x[node * IN_F + k] : (_Float16)0.f;
}

// W [K,Nout] f32 -> Bt [Nout,Kpad] f16 (transpose + pad)
__global__ void pack_bt(const float* __restrict__ W, _Float16* __restrict__ Bt,
                        int K, int Nout, int Kpad) {
    int i = blockIdx.x * blockDim.x + threadIdx.x;
    if (i >= Nout * Kpad) return;
    int nout = i / Kpad, k = i % Kpad;
    Bt[i] = (k < K) ? (_Float16)W[k * Nout + nout] : (_Float16)0.f;
}

// ---------------- WMMA GEMM: C[M,Nout] = A[M,Kpad] * Bt[Nout,Kpad]^T + bias ----------------
// one wave per 16x64 output strip (4 accumulators, A fragment reused 4x), 4 waves per block

__global__ __launch_bounds__(128)
void gemm_wmma(const _Float16* __restrict__ A, const _Float16* __restrict__ Bt,
               const float* __restrict__ bias, float* __restrict__ C,
               int M, int Nout, int Kpad) {
    int wave = threadIdx.x >> 5;
    int lane = threadIdx.x & 31;
    int tilesN = Nout >> 6;                      // strips of 64 columns
    int nTiles = (M >> 4) * tilesN;
    int tile = blockIdx.x * (blockDim.x >> 5) + wave;
    if (tile >= nTiles) return;                  // uniform per wave: EXEC stays all-ones
    int tileM = tile / tilesN, tileN = tile % tilesN;
    int l15 = lane & 15;
    bool hi = lane >= 16;

    // A fragment (16x32 f16): lane holds M=l15; K runs [koff..koff+15], koff=hi?16:0
    const _Float16* arow = A + (size_t)(tileM * 16 + l15) * Kpad + (hi ? 16 : 0);
    // B fragments (32x16 f16): lane holds N=l15; K runs [koff..koff+15], koff=hi?16:0 (column-contiguous in Bt)
    const _Float16* bcol = Bt + (size_t)(tileN * 64 + l15) * Kpad + (hi ? 16 : 0);
    const size_t bstride = (size_t)16 * Kpad;    // next 16-column tile of the strip

    v8f acc0 = {}, acc1 = {}, acc2 = {}, acc3 = {};
    for (int k0 = 0; k0 < Kpad; k0 += 32) {
        v8h a0 = *(const v8h*)(arow + k0);
        v8h a1 = *(const v8h*)(arow + k0 + 8);
        v16h a = __builtin_shufflevector(a0, a1, 0,1,2,3,4,5,6,7,8,9,10,11,12,13,14,15);

        v8h b0, b1;
        v16h b;
        b0 = *(const v8h*)(bcol + k0);
        b1 = *(const v8h*)(bcol + k0 + 8);
        b = __builtin_shufflevector(b0, b1, 0,1,2,3,4,5,6,7,8,9,10,11,12,13,14,15);
        acc0 = __builtin_amdgcn_wmma_f32_16x16x32_f16(false, a, false, b, (short)0, acc0, false, false);

        b0 = *(const v8h*)(bcol + bstride + k0);
        b1 = *(const v8h*)(bcol + bstride + k0 + 8);
        b = __builtin_shufflevector(b0, b1, 0,1,2,3,4,5,6,7,8,9,10,11,12,13,14,15);
        acc1 = __builtin_amdgcn_wmma_f32_16x16x32_f16(false, a, false, b, (short)0, acc1, false, false);

        b0 = *(const v8h*)(bcol + 2 * bstride + k0);
        b1 = *(const v8h*)(bcol + 2 * bstride + k0 + 8);
        b = __builtin_shufflevector(b0, b1, 0,1,2,3,4,5,6,7,8,9,10,11,12,13,14,15);
        acc2 = __builtin_amdgcn_wmma_f32_16x16x32_f16(false, a, false, b, (short)0, acc2, false, false);

        b0 = *(const v8h*)(bcol + 3 * bstride + k0);
        b1 = *(const v8h*)(bcol + 3 * bstride + k0 + 8);
        b = __builtin_shufflevector(b0, b1, 0,1,2,3,4,5,6,7,8,9,10,11,12,13,14,15);
        acc3 = __builtin_amdgcn_wmma_f32_16x16x32_f16(false, a, false, b, (short)0, acc3, false, false);
    }

    // C/D layout: lane -> col l15 (within 16-col tile), VGPR v -> row v + (hi?8:0)
    float* cbase = C + (size_t)(tileM * 16 + (hi ? 8 : 0)) * Nout;
    #pragma unroll
    for (int j = 0; j < 4; j++) {
        v8f acc = (j == 0) ? acc0 : (j == 1) ? acc1 : (j == 2) ? acc2 : acc3;
        int col = tileN * 64 + j * 16 + l15;
        float bv = bias[col];
        #pragma unroll
        for (int v = 0; v < 8; v++) cbase[(size_t)v * Nout + col] = acc[v] + bv;
    }
}

// ---------------- edge phase ----------------

// per (edge, head): score e = att_h . leaky(xl[src,h,:] + xr[dst,h,:]); segment max via ordered-uint atomicMax
template<int C>
__global__ void edge_score(const int* __restrict__ src, const int* __restrict__ dst,
                           const float* __restrict__ xl, const float* __restrict__ xr,
                           const float* __restrict__ att,
                           float* __restrict__ e, unsigned* __restrict__ m) {
    int i = blockIdx.x * blockDim.x + threadIdx.x;
    if (i >= N_EDGES * HEADS) return;
    int eidx = i / HEADS, h = i % HEADS;
    int s = src[eidx], t = dst[eidx];
    const float4* pl = (const float4*)(xl + (size_t)s * HEADS * C + h * C);
    const float4* pr = (const float4*)(xr + (size_t)t * HEADS * C + h * C);
    const float4* pa = (const float4*)(att + h * C);
    float acc = 0.f;
    for (int c = 0; c < C / 4; c++) {
        float4 a = pl[c], b = pr[c], w = pa[c];
        float z0 = a.x + b.x, z1 = a.y + b.y, z2 = a.z + b.z, z3 = a.w + b.w;
        z0 = z0 >= 0.f ? z0 : NEG_SLOPE * z0;
        z1 = z1 >= 0.f ? z1 : NEG_SLOPE * z1;
        z2 = z2 >= 0.f ? z2 : NEG_SLOPE * z2;
        z3 = z3 >= 0.f ? z3 : NEG_SLOPE * z3;
        acc += z0 * w.x + z1 * w.y + z2 * w.z + z3 * w.w;
    }
    e[i] = acc;
    atomicMax(&m[t * HEADS + h], f2o(acc));
}

// per (edge, head): w = exp(e - m[dst]); accumulate denom
__global__ void edge_weight(const int* __restrict__ dst, float* __restrict__ e,
                            const unsigned* __restrict__ m, float* __restrict__ denom) {
    int i = blockIdx.x * blockDim.x + threadIdx.x;
    if (i >= N_EDGES * HEADS) return;
    int eidx = i / HEADS, h = i % HEADS;
    int t = dst[eidx];
    float w = __expf(e[i] - o2f(m[t * HEADS + h]));
    e[i] = w;
    atomicAdd(&denom[t * HEADS + h], w);
}

__global__ void inv_denom(float* __restrict__ d, int n) {
    int i = blockIdx.x * blockDim.x + threadIdx.x;
    if (i < n) d[i] = 1.f / fmaxf(d[i], 1e-16f);
}

// per (edge, head, c/4): out[dst] += alpha * xl[src]
template<int C>
__global__ void edge_agg(const int* __restrict__ src, const int* __restrict__ dst,
                         const float* __restrict__ w, const float* __restrict__ rden,
                         const float* __restrict__ xl, float* __restrict__ out) {
    const int C4 = C / 4;
    int i = blockIdx.x * blockDim.x + threadIdx.x;
    if (i >= N_EDGES * HEADS * C4) return;
    int c4 = i % C4;
    int eh = i / C4;
    int eidx = eh / HEADS, h = eh % HEADS;
    int s = src[eidx], t = dst[eidx];
    float alpha = w[eh] * rden[t * HEADS + h];
    float4 v = ((const float4*)(xl + (size_t)s * HEADS * C + h * C))[c4];
    float* po = out + (size_t)t * HEADS * C + h * C + c4 * 4;
    atomicAdd(po + 0, alpha * v.x);
    atomicAdd(po + 1, alpha * v.y);
    atomicAdd(po + 2, alpha * v.z);
    atomicAdd(po + 3, alpha * v.w);
}

// ---------------- layer glue ----------------

// A2[N,224] f16 = [ x (15) | relu(agg1 + bias1) (192) | zeros (17) ]
__global__ void build_a2(const float* __restrict__ x, const float* __restrict__ agg1,
                         const float* __restrict__ bias1, _Float16* __restrict__ A2) {
    int i = blockIdx.x * blockDim.x + threadIdx.x;
    if (i >= N_NODES * K2PAD) return;
    int node = i / K2PAD, j = i % K2PAD;
    float v;
    if (j < IN_F) v = x[node * IN_F + j];
    else if (j < D2IN) {
        int hc = j - IN_F;
        float t = agg1[(size_t)node * HEADS * HID + hc] + bias1[hc];
        v = t > 0.f ? t : 0.f;
    } else v = 0.f;
    A2[i] = (_Float16)v;
}

// out[N,143] = [ x (15) | relu(mean_h(agg2) + bias2) (128) ]
__global__ void write_out(const float* __restrict__ x, const float* __restrict__ agg2,
                          const float* __restrict__ bias2, float* __restrict__ out) {
    const int OW = IN_F + OUT_FC;
    int i = blockIdx.x * blockDim.x + threadIdx.x;
    if (i >= N_NODES * OW) return;
    int node = i / OW, j = i % OW;
    float v;
    if (j < IN_F) v = x[node * IN_F + j];
    else {
        int c = j - IN_F;
        size_t base = (size_t)node * HEADS * OUT_FC;
        float t = (agg2[base + c] + agg2[base + OUT_FC + c] + agg2[base + 2 * OUT_FC + c])
                  * (1.f / 3.f) + bias2[c];
        v = t > 0.f ? t : 0.f;
    }
    out[i] = v;
}

// ---------------- launcher ----------------

static inline unsigned gdiv(long long n, int b) { return (unsigned)((n + b - 1) / b); }

extern "C" void kernel_launch(void* const* d_in, const int* in_sizes, int n_in,
                              void* d_out, int out_size, void* d_ws, size_t ws_size,
                              hipStream_t stream) {
    (void)in_sizes; (void)n_in; (void)out_size; (void)ws_size;
    const float* x     = (const float*)d_in[0];
    const int*   ei    = (const int*)d_in[1];
    const int*   src   = ei;
    const int*   dst   = ei + N_EDGES;
    const float* W1l   = (const float*)d_in[2];
    const float* b1l   = (const float*)d_in[3];
    const float* W1r   = (const float*)d_in[4];
    const float* b1r   = (const float*)d_in[5];
    const float* att1  = (const float*)d_in[6];
    const float* bias1 = (const float*)d_in[7];
    const float* W2l   = (const float*)d_in[8];
    const float* b2l   = (const float*)d_in[9];
    const float* W2r   = (const float*)d_in[10];
    const float* b2r   = (const float*)d_in[11];
    const float* att2  = (const float*)d_in[12];
    const float* bias2 = (const float*)d_in[13];

    char* ws = (char*)d_ws;
    size_t off = 0;
    auto alloc = [&](size_t bytes) { char* p = ws + off; off = (off + bytes + 255) & ~(size_t)255; return p; };
    float*    eBuf  = (float*)   alloc((size_t)N_EDGES * HEADS * 4);      // edge scores / weights
    unsigned* mBuf  = (unsigned*)alloc((size_t)N_NODES * HEADS * 4);      // segment max (ordered uint)
    float*    dBuf  = (float*)   alloc((size_t)N_NODES * HEADS * 4);      // denom -> 1/denom
    _Float16* Abuf  = (_Float16*)alloc((size_t)N_NODES * K2PAD * 2);      // f16 activations (A1 then A2)
    _Float16* BtL   = (_Float16*)alloc((size_t)384 * K2PAD * 2);          // Wl^T f16
    _Float16* BtR   = (_Float16*)alloc((size_t)384 * K2PAD * 2);          // Wr^T f16
    float*    xlBuf = (float*)   alloc((size_t)N_NODES * 384 * 4);
    float*    xrBuf = (float*)   alloc((size_t)N_NODES * 384 * 4);
    float*    agg   = (float*)   alloc((size_t)N_NODES * 384 * 4);

    const int B = 256;

    // ======== layer 1 ========
    pack_a1<<<gdiv((long long)N_NODES * K1PAD, B), B, 0, stream>>>(x, Abuf, N_NODES * K1PAD);
    pack_bt<<<gdiv(192 * K1PAD, B), B, 0, stream>>>(W1l, BtL, IN_F, 192, K1PAD);
    pack_bt<<<gdiv(192 * K1PAD, B), B, 0, stream>>>(W1r, BtR, IN_F, 192, K1PAD);
    {
        int tiles = (N_NODES / 16) * (192 / 64);
        int blocks = (tiles + 3) / 4;
        gemm_wmma<<<blocks, 128, 0, stream>>>(Abuf, BtL, b1l, xlBuf, N_NODES, 192, K1PAD);
        gemm_wmma<<<blocks, 128, 0, stream>>>(Abuf, BtR, b1r, xrBuf, N_NODES, 192, K1PAD);
    }
    fill_f32<<<gdiv(N_NODES * HEADS, B), B, 0, stream>>>((float*)mBuf, N_NODES * HEADS, 0.f);
    fill_f32<<<gdiv(N_NODES * HEADS, B), B, 0, stream>>>(dBuf, N_NODES * HEADS, 0.f);
    fill_f32<<<gdiv((long long)N_NODES * 192, B), B, 0, stream>>>(agg, N_NODES * 192, 0.f);
    edge_score<HID><<<gdiv((long long)N_EDGES * HEADS, B), B, 0, stream>>>(src, dst, xlBuf, xrBuf, att1, eBuf, mBuf);
    edge_weight<<<gdiv((long long)N_EDGES * HEADS, B), B, 0, stream>>>(dst, eBuf, mBuf, dBuf);
    inv_denom<<<gdiv(N_NODES * HEADS, B), B, 0, stream>>>(dBuf, N_NODES * HEADS);
    edge_agg<HID><<<gdiv((long long)N_EDGES * HEADS * (HID / 4), B), B, 0, stream>>>(src, dst, eBuf, dBuf, xlBuf, agg);
    build_a2<<<gdiv((long long)N_NODES * K2PAD, B), B, 0, stream>>>(x, agg, bias1, Abuf);

    // ======== layer 2 ========
    pack_bt<<<gdiv(384 * K2PAD, B), B, 0, stream>>>(W2l, BtL, D2IN, 384, K2PAD);
    pack_bt<<<gdiv(384 * K2PAD, B), B, 0, stream>>>(W2r, BtR, D2IN, 384, K2PAD);
    {
        int tiles = (N_NODES / 16) * (384 / 64);
        int blocks = (tiles + 3) / 4;
        gemm_wmma<<<blocks, 128, 0, stream>>>(Abuf, BtL, b2l, xlBuf, N_NODES, 384, K2PAD);
        gemm_wmma<<<blocks, 128, 0, stream>>>(Abuf, BtR, b2r, xrBuf, N_NODES, 384, K2PAD);
    }
    fill_f32<<<gdiv(N_NODES * HEADS, B), B, 0, stream>>>((float*)mBuf, N_NODES * HEADS, 0.f);
    fill_f32<<<gdiv(N_NODES * HEADS, B), B, 0, stream>>>(dBuf, N_NODES * HEADS, 0.f);
    fill_f32<<<gdiv((long long)N_NODES * 384, B), B, 0, stream>>>(agg, N_NODES * 384, 0.f);
    edge_score<OUT_FC><<<gdiv((long long)N_EDGES * HEADS, B), B, 0, stream>>>(src, dst, xlBuf, xrBuf, att2, eBuf, mBuf);
    edge_weight<<<gdiv((long long)N_EDGES * HEADS, B), B, 0, stream>>>(dst, eBuf, mBuf, dBuf);
    inv_denom<<<gdiv(N_NODES * HEADS, B), B, 0, stream>>>(dBuf, N_NODES * HEADS);
    edge_agg<OUT_FC><<<gdiv((long long)N_EDGES * HEADS * (OUT_FC / 4), B), B, 0, stream>>>(src, dst, eBuf, dBuf, xlBuf, agg);
    write_out<<<gdiv((long long)N_NODES * (IN_F + OUT_FC), B), B, 0, stream>>>(x, agg, bias2, (float*)d_out);
}
